// GraphEncoder_4458176053822
// MI455X (gfx1250) — compile-verified, matrix operand order checked by reading, not run on hardware
//
#include <hip/hip_runtime.h>
#include <hip/hip_bf16.h>

#define DEV __device__ __forceinline__

typedef __attribute__((ext_vector_type(16))) short    v16s;
typedef __attribute__((ext_vector_type(16))) __bf16   v16bf;
typedef __attribute__((ext_vector_type(8)))  float    v8f;
typedef __attribute__((ext_vector_type(8)))  short    v8s;

// ---------------------------------------------------------------- helpers
DEV unsigned short f2bf(float f) {
  unsigned int u = __float_as_uint(f);
  return (unsigned short)((u + 0x7fffu + ((u >> 16) & 1u)) >> 16);  // RNE
}

DEV float gelu_exact(float x) { return 0.5f * x * (1.f + erff(x * 0.70710678118654752f)); }

// A/B fragment for v_wmma_f32_16x16x32_bf16 (16 bf16 per lane):
// element j holds K = kk + (j>>3)*16 + half*8 + (j&7)   (half = lane>>4)
// caller passes p already offset by  line*ld + kk + half*8.
DEV v16s frag_from_f32(const float* p) {
  const float4* p4 = (const float4*)p;           // 16B-aligned by construction
  float4 x0 = p4[0], x1 = p4[1], x2 = p4[4], x3 = p4[5];
  v16s r;
  r[0]=(short)f2bf(x0.x);  r[1]=(short)f2bf(x0.y);  r[2]=(short)f2bf(x0.z);  r[3]=(short)f2bf(x0.w);
  r[4]=(short)f2bf(x1.x);  r[5]=(short)f2bf(x1.y);  r[6]=(short)f2bf(x1.z);  r[7]=(short)f2bf(x1.w);
  r[8]=(short)f2bf(x2.x);  r[9]=(short)f2bf(x2.y);  r[10]=(short)f2bf(x2.z); r[11]=(short)f2bf(x2.w);
  r[12]=(short)f2bf(x3.x); r[13]=(short)f2bf(x3.y); r[14]=(short)f2bf(x3.z); r[15]=(short)f2bf(x3.w);
  return r;
}

DEV v16s frag_from_bf16(const unsigned short* p) {
  v8s a = *(const v8s*)p;          // K rel 0..7
  v8s b = *(const v8s*)(p + 16);   // K rel 16..23
  v16s r;
#pragma unroll
  for (int i = 0; i < 8; ++i) { r[i] = a[i]; r[8 + i] = b[i]; }
  return r;
}

// B stored (K,N) row-major (non-transposed GEMM): element (k, col), stride ld.
DEV v16s frag_from_bf16_strided(const unsigned short* p, long ld) {
  v16s r;
#pragma unroll
  for (int j = 0; j < 8; ++j) r[j] = (short)p[(long)j * ld];
  const unsigned short* q = p + 16 * ld;
#pragma unroll
  for (int j = 0; j < 8; ++j) r[8 + j] = (short)q[(long)j * ld];
  return r;
}

DEV v8f wmma_bf16(v16s a, v16s b, v8f c) {
  return __builtin_amdgcn_wmma_f32_16x16x32_bf16(
      false, __builtin_bit_cast(v16bf, a),
      false, __builtin_bit_cast(v16bf, b),
      (short)0, c, false, false);
}

// ---------------------------------------------------------------- generic batched GEMM
// C[z,m,n] = act( (sum_k A[z,m,k]*B[z,(n,k) or (k,n)] + bias[n]) * scale )
// Workgroup: 8 waves -> 32x64 output tile, wave = one 16x16 WMMA tile.
template <bool ABF, bool BBF, bool TB, bool OBF, int ACT>
__global__ __launch_bounds__(256) void wmma_gemm_kernel(
    const void* __restrict__ Ap, long lda, long sAz,
    const void* __restrict__ Bp, long ldb, long sBz,
    const float* __restrict__ bias, float scale,
    void* __restrict__ Cp, long ldc, long sCz,
    int M, int N, int K) {
  const int w = threadIdx.x >> 5;
  const int lane = threadIdx.x & 31;
  const int half = lane >> 4;
  const int lr = lane & 15;
  const int m0 = blockIdx.y * 32 + (w & 1) * 16;
  const int n0 = blockIdx.x * 64 + (w >> 1) * 16;
  if (m0 >= M || n0 >= N) return;  // wave-uniform; EXEC stays all-ones for WMMA
  const long zA = (long)blockIdx.z * sAz;
  const long zB = (long)blockIdx.z * sBz;
  const long zC = (long)blockIdx.z * sCz;

  v8f acc;
#pragma unroll
  for (int i = 0; i < 8; ++i) acc[i] = 0.f;

  for (int kk = 0; kk < K; kk += 32) {
    v16s af, bfr;
    if (ABF) af = frag_from_bf16((const unsigned short*)Ap + zA + (long)(m0 + lr) * lda + kk + half * 8);
    else     af = frag_from_f32((const float*)Ap + zA + (long)(m0 + lr) * lda + kk + half * 8);
    if (TB) {
      if (BBF) bfr = frag_from_bf16((const unsigned short*)Bp + zB + (long)(n0 + lr) * ldb + kk + half * 8);
      else     bfr = frag_from_f32((const float*)Bp + zB + (long)(n0 + lr) * ldb + kk + half * 8);
    } else {
      bfr = frag_from_bf16_strided((const unsigned short*)Bp + zB + (long)(kk + half * 8) * ldb + (n0 + lr), ldb);
    }
    acc = wmma_bf16(af, bfr, acc);
  }

  const int n = n0 + lr;
  const float bv = bias ? bias[n] : 0.f;
#pragma unroll
  for (int r = 0; r < 8; ++r) {
    const int m = m0 + half * 8 + r;   // C/D layout: m = half*8 + r, n = lane&15
    float v = (acc[r] + bv) * scale;
    if (ACT == 1) v = gelu_exact(v);
    const long idx = zC + (long)m * ldc + n;
    if (OBF) ((unsigned short*)Cp)[idx] = f2bf(v);
    else     ((float*)Cp)[idx] = v;
  }
}

template <bool ABF, bool BBF, bool TB, bool OBF, int ACT>
static void launch_gemm(hipStream_t st, const void* A, long lda, long sAz,
                        const void* Bm, long ldb, long sBz,
                        const float* bias, float scale,
                        void* C, long ldc, long sCz, int M, int N, int K, int Z) {
  dim3 g((unsigned)((N + 63) / 64), (unsigned)((M + 31) / 32), (unsigned)Z);
  wmma_gemm_kernel<ABF, BBF, TB, OBF, ACT><<<g, dim3(256, 1, 1), 0, st>>>(
      A, lda, sAz, Bm, ldb, sBz, bias, scale, C, ldc, sCz, M, N, K);
}

// ---------------------------------------------------------------- graph softmax
// rows = NH*L = 6144, cols = 512. One wave per row. p = bf16(softmax(s+extra+mask)).
__global__ __launch_bounds__(256) void graph_softmax_kernel(
    const float* __restrict__ s, const float* __restrict__ extra,
    const float* __restrict__ mask, unsigned short* __restrict__ p) {
  const int row = blockIdx.x * 8 + (threadIdx.x >> 5);
  const int lane = threadIdx.x & 31;
  const float* sr = s + (long)row * 512;
  const float* er = extra + (long)row * 512;
  const float* mr = mask + (long)(row & 511) * 512;  // mask broadcast over heads
  float v[16];
  float mx = -3.0e38f;
#pragma unroll
  for (int i = 0; i < 16; ++i) {
    const int c = lane + i * 32;
    v[i] = sr[c] + er[c] + mr[c];
    mx = fmaxf(mx, v[i]);
  }
#pragma unroll
  for (int o = 16; o > 0; o >>= 1) mx = fmaxf(mx, __shfl_xor(mx, o, 32));
  float sum = 0.f;
#pragma unroll
  for (int i = 0; i < 16; ++i) { v[i] = __expf(v[i] - mx); sum += v[i]; }
#pragma unroll
  for (int o = 16; o > 0; o >>= 1) sum += __shfl_xor(sum, o, 32);
  const float inv = 1.f / sum;
  unsigned short* pr = p + (long)row * 512;
#pragma unroll
  for (int i = 0; i < 16; ++i) pr[lane + i * 32] = f2bf(v[i] * inv);
}

// ---------------------------------------------------------------- residual + LayerNorm (cols = 768)
__global__ __launch_bounds__(256) void add_ln_kernel(
    const float* __restrict__ x, const float* __restrict__ res,
    const float* __restrict__ g, const float* __restrict__ b,
    float* __restrict__ y) {
  __shared__ float red[256];
  const int row = blockIdx.x, t = threadIdx.x;
  const float* xr = x + (long)row * 768;
  const float* rr = res + (long)row * 768;
  float v[3];
#pragma unroll
  for (int i = 0; i < 3; ++i) v[i] = xr[t + i * 256] + rr[t + i * 256];
  red[t] = v[0] + v[1] + v[2];
  __syncthreads();
  for (int o = 128; o > 0; o >>= 1) { if (t < o) red[t] += red[t + o]; __syncthreads(); }
  const float mean = red[0] * (1.f / 768.f);
  __syncthreads();
  float q = 0.f;
#pragma unroll
  for (int i = 0; i < 3; ++i) { const float d = v[i] - mean; q += d * d; }
  red[t] = q;
  __syncthreads();
  for (int o = 128; o > 0; o >>= 1) { if (t < o) red[t] += red[t + o]; __syncthreads(); }
  const float inv = rsqrtf(red[0] * (1.f / 768.f) + 1e-12f);
#pragma unroll
  for (int i = 0; i < 3; ++i) {
    const int c = t + i * 256;
    y[(long)row * 768 + c] = (v[i] - mean) * inv * g[c] + b[c];
  }
}

// ---------------------------------------------------------------- fused cross-attn scores + softmax
// grid = (S/16, B*CH). Workgroup: 16 queries x 512 labels; 8 waves x (4 WMMA n-tiles), K=96.
// Scores staged in LDS (16x512 f32 = 32 KB), row softmax, single write of the 256 MB output.
__global__ __launch_bounds__(256) void cross_attn_softmax_kernel(
    const unsigned short* __restrict__ qbf,  // (B*S, 768) bf16, pre-scaled by hd^-0.5
    const unsigned short* __restrict__ kbf,  // (L, 768)  bf16
    const float* __restrict__ mask,          // (B, S, L) f32
    float* __restrict__ out) {               // (B, CH, S, L) f32
  __shared__ float smem[16 * 512];
  const int w = threadIdx.x >> 5;
  const int lane = threadIdx.x & 31;
  const int half = lane >> 4;
  const int lr = lane & 15;
  const int s0 = blockIdx.x * 16;
  const int b = blockIdx.y >> 3;
  const int h = blockIdx.y & 7;

  const unsigned short* Abase = qbf + ((long)(b * 1024 + s0)) * 768 + h * 96;
  const unsigned short* Bbase = kbf + h * 96;

  v8f acc[4];
#pragma unroll
  for (int t = 0; t < 4; ++t)
#pragma unroll
    for (int i = 0; i < 8; ++i) acc[t][i] = 0.f;

  for (int kk = 0; kk < 96; kk += 32) {
    const v16s af = frag_from_bf16(Abase + (long)lr * 768 + kk + half * 8);
#pragma unroll
    for (int t = 0; t < 4; ++t) {
      const int n = w * 64 + t * 16 + lr;
      const v16s bfr = frag_from_bf16(Bbase + (long)n * 768 + kk + half * 8);
      acc[t] = wmma_bf16(af, bfr, acc[t]);
    }
  }

  const float* mrow = mask + ((long)(b * 1024 + s0)) * 512;
#pragma unroll
  for (int t = 0; t < 4; ++t) {
    const int n = w * 64 + t * 16 + lr;
#pragma unroll
    for (int r = 0; r < 8; ++r) {
      const int m = half * 8 + r;
      smem[m * 512 + n] = acc[t][r] + mrow[(long)m * 512 + n];
    }
  }
  __syncthreads();

  // each wave softmaxes 2 rows of 512
#pragma unroll
  for (int rr = 0; rr < 2; ++rr) {
    const int m = w * 2 + rr;
    const float* row = smem + m * 512;
    float v[16];
    float mx = -3.0e38f;
#pragma unroll
    for (int i = 0; i < 16; ++i) { v[i] = row[lane + i * 32]; mx = fmaxf(mx, v[i]); }
#pragma unroll
    for (int o = 16; o > 0; o >>= 1) mx = fmaxf(mx, __shfl_xor(mx, o, 32));
    float sum = 0.f;
#pragma unroll
    for (int i = 0; i < 16; ++i) { v[i] = __expf(v[i] - mx); sum += v[i]; }
#pragma unroll
    for (int o = 16; o > 0; o >>= 1) sum += __shfl_xor(sum, o, 32);
    const float inv = 1.f / sum;
    float* orow = out + (((long)(b * 8 + h) * 1024) + s0 + m) * 512;
#pragma unroll
    for (int i = 0; i < 16; ++i) orow[lane + i * 32] = v[i] * inv;
  }
}

// ---------------------------------------------------------------- host driver
extern "C" void kernel_launch(void* const* d_in, const int* in_sizes, int n_in,
                              void* d_out, int out_size, void* d_ws, size_t ws_size,
                              hipStream_t stream) {
  (void)in_sizes; (void)n_in; (void)out_size; (void)ws_size;
  const int H = 768, NH = 12, CHh = 8, IM = 3072, L = 512, B = 16, S = 1024;

  const float* label = (const float*)d_in[0];
  const float* extra = (const float*)d_in[1];
  const float* smask = (const float*)d_in[2];
  const float* xemb  = (const float*)d_in[3];
  const float* cmask = (const float*)d_in[4];
  // params flattened: cross{kb,kw,qb,qw}, l0{b1,b2,kb,kw,ln2_b,ln2_g,ln_b,ln_g,ob,ow,qb,qw,vb,vw,w1,w2},
  //                   l1{kb,kw,ln_b,ln_g,ob,ow,qb,qw,vb,vw}
  const float* c_kb = (const float*)d_in[5];
  const float* c_kw = (const float*)d_in[6];
  const float* c_qb = (const float*)d_in[7];
  const float* c_qw = (const float*)d_in[8];
  const float* l0_b1   = (const float*)d_in[9];
  const float* l0_b2   = (const float*)d_in[10];
  const float* l0_kb   = (const float*)d_in[11];
  const float* l0_kw   = (const float*)d_in[12];
  const float* l0_ln2b = (const float*)d_in[13];
  const float* l0_ln2g = (const float*)d_in[14];
  const float* l0_lnb  = (const float*)d_in[15];
  const float* l0_lng  = (const float*)d_in[16];
  const float* l0_ob   = (const float*)d_in[17];
  const float* l0_ow   = (const float*)d_in[18];
  const float* l0_qb   = (const float*)d_in[19];
  const float* l0_qw   = (const float*)d_in[20];
  const float* l0_vb   = (const float*)d_in[21];
  const float* l0_vw   = (const float*)d_in[22];
  const float* l0_w1   = (const float*)d_in[23];
  const float* l0_w2   = (const float*)d_in[24];
  const float* l1_kb   = (const float*)d_in[25];
  const float* l1_kw   = (const float*)d_in[26];
  const float* l1_lnb  = (const float*)d_in[27];
  const float* l1_lng  = (const float*)d_in[28];
  const float* l1_ob   = (const float*)d_in[29];
  const float* l1_ow   = (const float*)d_in[30];
  const float* l1_qb   = (const float*)d_in[31];
  const float* l1_qw   = (const float*)d_in[32];
  const float* l1_vb   = (const float*)d_in[33];
  const float* l1_vw   = (const float*)d_in[34];

  char* wp = (char*)d_ws;
  auto alloc = [&](size_t bytes) -> void* {
    void* r = (void*)wp;
    wp += (bytes + 255) & ~(size_t)255;
    return r;
  };
  float*          hA  = (float*)alloc((size_t)L * H * 4);
  float*          hB  = (float*)alloc((size_t)L * H * 4);
  unsigned short* qg  = (unsigned short*)alloc((size_t)L * H * 2);
  unsigned short* kg  = (unsigned short*)alloc((size_t)L * H * 2);
  unsigned short* vg  = (unsigned short*)alloc((size_t)L * H * 2);
  float*          sc  = (float*)alloc((size_t)NH * L * L * 4);
  unsigned short* pbf = (unsigned short*)alloc((size_t)NH * L * L * 2);
  float*          ao  = (float*)alloc((size_t)L * H * 4);
  float*          tmp = (float*)alloc((size_t)L * IM * 4);
  unsigned short* qc  = (unsigned short*)alloc((size_t)B * S * H * 2);
  unsigned short* kc  = (unsigned short*)alloc((size_t)L * H * 2);

  auto sa_block = [&](const float* hin,
                      const float* qw, const float* qb, const float* kw, const float* kb,
                      const float* vw, const float* vb, const float* ow, const float* ob,
                      const float* lng, const float* lnb, float* hout) {
    // q/k/v projections -> bf16 (q pre-scaled by hd^-0.5 = 1/8)
    launch_gemm<false, false, true, true, 0>(stream, hin, H, 0, qw, H, 0, qb, 0.125f, qg, H, 0, L, H, H, 1);
    launch_gemm<false, false, true, true, 0>(stream, hin, H, 0, kw, H, 0, kb, 1.f,    kg, H, 0, L, H, H, 1);
    launch_gemm<false, false, true, true, 0>(stream, hin, H, 0, vw, H, 0, vb, 1.f,    vg, H, 0, L, H, H, 1);
    // per-head scores: s[z] = q_z @ k_z^T  (z = 12 heads, K = 64)
    launch_gemm<true, true, true, false, 0>(stream, qg, H, 64, kg, H, 64, nullptr, 1.f,
                                            sc, L, (long)L * L, L, L, 64, NH);
    graph_softmax_kernel<<<dim3(NH * L / 8), dim3(256), 0, stream>>>(sc, extra, smask, pbf);
    // per-head A @ V  (K = 512, N = 64), output interleaved back to (L, H)
    launch_gemm<true, true, false, false, 0>(stream, pbf, L, (long)L * L, vg, H, 64, nullptr, 1.f,
                                             ao, H, 64, L, 64, L, NH);
    // output projection
    launch_gemm<false, false, true, false, 0>(stream, ao, H, 0, ow, H, 0, ob, 1.f, tmp, H, 0, L, H, H, 1);
    add_ln_kernel<<<dim3(L), dim3(256), 0, stream>>>(tmp, hin, lng, lnb, hout);
  };

  // GraphLayer 0: self-attn block + FFN block
  sa_block(label, l0_qw, l0_qb, l0_kw, l0_kb, l0_vw, l0_vb, l0_ow, l0_ob, l0_lng, l0_lnb, hA);
  launch_gemm<false, false, true, false, 1>(stream, hA, H, 0, l0_w1, H, 0, l0_b1, 1.f, tmp, IM, 0, L, IM, H, 1);
  launch_gemm<false, false, true, false, 0>(stream, tmp, IM, 0, l0_w2, IM, 0, l0_b2, 1.f, ao, H, 0, L, H, IM, 1);
  add_ln_kernel<<<dim3(L), dim3(256), 0, stream>>>(ao, hA, l0_ln2g, l0_ln2b, hB);

  // GraphLayer 1: self-attn block
  sa_block(hB, l1_qw, l1_qb, l1_kw, l1_kb, l1_vw, l1_vb, l1_ow, l1_ob, l1_lng, l1_lnb, hA);

  // Cross-attention: q = (xemb @ qw^T + qb) * 96^-0.5 -> bf16 ; k = h @ kw^T + kb -> bf16
  launch_gemm<false, false, true, true, 0>(stream, xemb, H, 0, c_qw, H, 0, c_qb, 0.1020620726159658f,
                                           qc, H, 0, B * S, H, H, 1);
  launch_gemm<false, false, true, true, 0>(stream, hA, H, 0, c_kw, H, 0, c_kb, 1.f, kc, H, 0, L, H, H, 1);

  cross_attn_softmax_kernel<<<dim3(S / 16, B * CHh), dim3(256), 0, stream>>>(
      qc, kc, cmask, (float*)d_out);
}
